// Net_60370060313335
// MI455X (gfx1250) — compile-verified, matrix operand order checked by reading, not run on hardware
//
#include <hip/hip_runtime.h>
#include <stdint.h>

typedef __attribute__((ext_vector_type(16))) _Float16 v16h;
typedef __attribute__((ext_vector_type(8)))  _Float16 v8h;
typedef __attribute__((ext_vector_type(8)))  float    v8f;

#define BN_EPS 1e-5f

// Deterministic counter-based hash RNG -> uniform [0,1). Replaces JAX threefry
// (statistically equivalent; exact stream is unobservable here).
__device__ __forceinline__ float urand(uint32_t a, uint32_t b) {
  uint32_t h = a * 0x9E3779B9u + 0x7F4A7C15u;
  h ^= b * 0x85EBCA6Bu;
  h ^= h >> 16; h *= 0x7FEB352Du;
  h ^= h >> 15; h *= 0x846CA68Bu;
  h ^= h >> 16;
  return (float)(h >> 8) * (1.0f / 16777216.0f);
}

// ---------------------------------------------------------------------------
// Effective binarized weights: mean of NPASS=4 stochastic sign draws.
// kind 0: conv OIHW -> [n][tap*8 + ci]  (channel-padded NHWC im2col order,
//         tap = ky*5+kx, pad: ci>=Cin or tap>=25 or n>=Nv -> 0)
// kind 1: fc, plain [n][k]
// kind 2: fc1, permute CHW-flatten -> HWC-flatten ([n][p*16+ci] <- src[n][ci*25+p])
// ---------------------------------------------------------------------------
__global__ void k_make_w(const float* __restrict__ src, _Float16* __restrict__ dst,
                         int Nv, int Npad, int Kv, int Kpad, int kind, int Cin,
                         uint32_t seed) {
  int idx = blockIdx.x * 256 + threadIdx.x;
  if (idx >= Npad * Kpad) return;
  int n = idx / Kpad, k = idx % Kpad;
  bool ok = (n < Nv);
  int si = 0;
  if (kind == 0) {
    int tap = k >> 3, ci = k & 7, KHW = Kv / Cin;   // KHW == 25
    ok = ok && (ci < Cin) && (tap < KHW);
    si = (n * Cin + ci) * KHW + tap;
  } else if (kind == 2) {
    int p = k / 16, ci = k % 16;
    ok = ok && (k < Kv);
    si = n * Kv + ci * 25 + p;
  } else {
    ok = ok && (k < Kv);
    si = n * Kv + k;
  }
  _Float16 o = (_Float16)0.f;
  if (ok) {
    float w  = src[si];
    float pr = fminf(fmaxf((w + 1.f) * 0.5f, 0.f), 1.f);
    float acc = 0.f;
#pragma unroll
    for (int pass = 0; pass < 4; ++pass)
      acc += (urand(seed + pass, (uint32_t)si) < pr) ? 1.f : -1.f;
    o = (_Float16)(acc * 0.25f);   // {-1,-0.5,0,0.5,1}: exact in f16
  }
  dst[idx] = o;
}

// NCHW f32 -> NHWC f16 with channels padded 3->8 (zeros)
__global__ void k_x2h(const float* __restrict__ x, _Float16* __restrict__ xh, int total) {
  int idx = blockIdx.x * 256 + threadIdx.x;
  if (idx >= total) return;
  int c = idx & 7, t = idx >> 3;
  int px = t & 31; t >>= 5;
  int py = t & 31; int b = t >> 5;
  xh[idx] = (c < 3) ? (_Float16)x[((b * 3 + c) * 32 + py) * 32 + px] : (_Float16)0.f;
}

// ---------------------------------------------------------------------------
// Fragment swizzle per ISA 7.12.2 (16-bit A/B): element i of v16h holds
// K = kk + (i>>3)*16 + half*8 + (i&7), half = lane>>4.
// ---------------------------------------------------------------------------

// FC GEMM: one wave computes a 16x16 tile of C = A(MxK) * W^T, W is [Npad][Kpad].
template <typename OutT>
__global__ __launch_bounds__(128)
void k_gemm_fc(const _Float16* __restrict__ A, const _Float16* __restrict__ W,
               OutT* __restrict__ C, int Kpad, int Astride, int Nstride) {
  const int lane = threadIdx.x & 31;
  const int wave = threadIdx.x >> 5;
  const int row0 = (blockIdx.x * 4 + wave) * 16;
  const int col0 = blockIdx.y * 16;
  const int half = (lane >> 4) & 1;
  const int mn   = lane & 15;

  v8f acc = {};
  for (int kk = 0; kk < Kpad; kk += 32) {
    v16h bfrag;
    {
      const _Float16* wr = W + (size_t)(col0 + mn) * Kpad + kk + half * 8;
      v8h lo = *(const v8h*)wr;
      v8h hi = *(const v8h*)(wr + 16);
      __builtin_prefetch(wr + 32, 0, 1);
#pragma unroll
      for (int i = 0; i < 8; ++i) { bfrag[i] = lo[i]; bfrag[8 + i] = hi[i]; }
    }
    v16h afrag;
    {
      const _Float16* ar = A + (size_t)(row0 + mn) * Astride + kk + half * 8;
      v8h lo = *(const v8h*)ar;
      v8h hi = *(const v8h*)(ar + 16);
#pragma unroll
      for (int i = 0; i < 8; ++i) { afrag[i] = lo[i]; afrag[8 + i] = hi[i]; }
    }
    acc = __builtin_amdgcn_wmma_f32_16x16x32_f16(false, afrag, false, bfrag,
                                                 (short)0, acc, false, false);
  }
  OutT* crow = C + (size_t)(row0 + half * 8) * Nstride + col0 + mn;
#pragma unroll
  for (int r = 0; r < 8; ++r) crow[(size_t)r * Nstride] = (OutT)acc[r];
}

// Conv implicit-im2col GEMM on channel-padded NHWC (8 ch): k = tap*8 + ci,
// so each 8-element fragment run is one tap = one aligned 16B global_load_b128.
// KPAD = 28 taps * 8 = 224; taps 25..27 and pad channels hit zero weights.
// Tap = C + half with C compile-time after unroll, so both candidate image
// offsets fold to literals and selection is a single cndmask on `half`.
// Weights [16][224]; C is [M][16] f16. KW=KH=5.
template <int HIN, int WIN, int HOUT, int WOUT>
__global__ __launch_bounds__(128)
void k_gemm_conv(const _Float16* __restrict__ A, const _Float16* __restrict__ W,
                 _Float16* __restrict__ C) {
  constexpr int KPAD = 224;
  const int lane = threadIdx.x & 31;
  const int wave = threadIdx.x >> 5;
  const int row0 = (blockIdx.x * 4 + wave) * 16;
  const int half = (lane >> 4) & 1;
  const int mn   = lane & 15;

  // row -> (b, oy, ox); constant divisors -> magic multiply (once per kernel)
  const int m    = row0 + mn;
  const int bimg = m / (HOUT * WOUT);
  const int pos  = m % (HOUT * WOUT);
  const int oy   = pos / WOUT, ox = pos % WOUT;
  const _Float16* base =
      A + (size_t)bimg * (HIN * WIN * 8) + (oy * WIN + ox) * 8;

  v8f acc = {};
#pragma unroll
  for (int kk = 0; kk < KPAD; kk += 32) {
    v16h bfrag;
    {
      const _Float16* wr = W + (size_t)mn * KPAD + kk + half * 8;
      v8h lo = *(const v8h*)wr;
      v8h hi = *(const v8h*)(wr + 16);
      __builtin_prefetch(wr + 32, 0, 1);
#pragma unroll
      for (int i = 0; i < 8; ++i) { bfrag[i] = lo[i]; bfrag[8 + i] = hi[i]; }
    }
    v16h afrag;
#pragma unroll
    for (int r = 0; r < 2; ++r) {
      const int Ct = (kk >> 3) + r * 2;                       // literal after unroll
      const int o0 = ((Ct / 5) * WIN + (Ct % 5)) * 8;         // literal
      const int o1 = (((Ct + 1) / 5) * WIN + ((Ct + 1) % 5)) * 8; // literal
      v8h v = {};
      if (Ct + 1 < 25) {                 // compile-time: both taps valid
        v = *(const v8h*)(base + (half ? o1 : o0));
      } else if (Ct < 25) {              // compile-time reach; runtime half test
        if (!half) v = *(const v8h*)(base + o0);
      }                                  // else: pad taps, stay zero
#pragma unroll
      for (int i = 0; i < 8; ++i) afrag[r * 8 + i] = v[i];
    }
    acc = __builtin_amdgcn_wmma_f32_16x16x32_f16(false, afrag, false, bfrag,
                                                 (short)0, acc, false, false);
  }
  _Float16* crow = C + (size_t)(row0 + half * 8) * 16 + mn;
#pragma unroll
  for (int r = 0; r < 8; ++r) crow[(size_t)r * 16] = (_Float16)acc[r];
}

// 2x2 max pool over NHWC f16, channel stride Cs preserved
__global__ void k_pool(const _Float16* __restrict__ in, _Float16* __restrict__ out,
                       int B, int H, int Wd, int Cs) {
  int Hp = H >> 1, Wp = Wd >> 1;
  long total = (long)B * Hp * Wp * Cs;
  long idx = (long)blockIdx.x * 256 + threadIdx.x;
  if (idx >= total) return;
  int c = idx % Cs; long t = idx / Cs;
  int px = t % Wp; t /= Wp;
  int py = t % Hp; int b = t / Hp;
  long base = (((long)b * H + py * 2) * Wd + px * 2) * Cs + c;
  float v0 = (float)in[base],                 v1 = (float)in[base + Cs];
  float v2 = (float)in[base + (long)Wd * Cs], v3 = (float)in[base + (long)Wd * Cs + Cs];
  out[idx] = (_Float16)fmaxf(fmaxf(v0, v1), fmaxf(v2, v3));
}

// Per-channel sum / sum-of-squares: one block per channel, LDS tree reduce
template <typename T>
__global__ void k_stat(const T* __restrict__ x, float* __restrict__ stats,
                       int stride, long count) {
  int c = blockIdx.x;
  __shared__ float s0[256], s1[256];
  float sum = 0.f, sq = 0.f;
  for (long i = threadIdx.x; i < count; i += 256) {
    float v = (float)x[i * stride + c];
    sum += v; sq += v * v;
  }
  s0[threadIdx.x] = sum; s1[threadIdx.x] = sq;
  __syncthreads();
  for (int s = 128; s > 0; s >>= 1) {
    if ((int)threadIdx.x < s) {
      s0[threadIdx.x] += s0[threadIdx.x + s];
      s1[threadIdx.x] += s1[threadIdx.x + s];
    }
    __syncthreads();
  }
  if (threadIdx.x == 0) { stats[2 * c] = s0[0]; stats[2 * c + 1] = s1[0]; }
}

// BN + stochastic sign (forward of _stanh is just +-1); pad columns -> 0
template <typename T>
__global__ void k_bn_act(const T* __restrict__ src, int sstride,
                         const float* __restrict__ stats,
                         const float* __restrict__ g, const float* __restrict__ b,
                         _Float16* __restrict__ dst, int dstride, int cvalid,
                         long rows, float invcnt, uint32_t seed) {
  long total = rows * dstride;
  long idx = (long)blockIdx.x * 256 + threadIdx.x;
  if (idx >= total) return;
  long row = idx / dstride; int c = idx % dstride;
  _Float16 o = (_Float16)0.f;
  if (c < cvalid) {
    float m   = stats[2 * c] * invcnt;
    float var = stats[2 * c + 1] * invcnt - m * m;
    float y = ((float)src[row * sstride + c] - m) * rsqrtf(var + BN_EPS) * g[c] + b[c];
    float p = fminf(fmaxf((y + 1.f) * 0.5f, 0.f), 1.f);
    o = (urand(seed, (uint32_t)idx) < p) ? (_Float16)1.f : (_Float16)(-1.f);
  }
  dst[idx] = o;
}

// Final BN -> f32 output [8192,10]
__global__ void k_bn_out(const float* __restrict__ src, int sstride,
                         const float* __restrict__ stats,
                         const float* __restrict__ g, const float* __restrict__ b,
                         float* __restrict__ dst, int dstride, long rows, float invcnt) {
  long total = rows * dstride;
  long idx = (long)blockIdx.x * 256 + threadIdx.x;
  if (idx >= total) return;
  long row = idx / dstride; int c = idx % dstride;
  float m   = stats[2 * c] * invcnt;
  float var = stats[2 * c + 1] * invcnt - m * m;
  dst[idx] = ((float)src[row * sstride + c] - m) * rsqrtf(var + BN_EPS) * g[c] + b[c];
}

// ---------------------------------------------------------------------------
extern "C" void kernel_launch(void* const* d_in, const int* in_sizes, int n_in,
                              void* d_out, int out_size, void* d_ws, size_t ws_size,
                              hipStream_t stream) {
  (void)in_sizes; (void)n_in; (void)out_size; (void)ws_size;
  const float* x       = (const float*)d_in[0];
  const float* conv1_w = (const float*)d_in[1];
  const float* conv2_w = (const float*)d_in[2];
  const float* fc1_w   = (const float*)d_in[3];
  const float* fc2_w   = (const float*)d_in[4];
  const float* fc3_w   = (const float*)d_in[5];
  const float* bn_g[5] = { (const float*)d_in[6], (const float*)d_in[8],
                           (const float*)d_in[10], (const float*)d_in[12],
                           (const float*)d_in[14] };
  const float* bn_b[5] = { (const float*)d_in[7], (const float*)d_in[9],
                           (const float*)d_in[11], (const float*)d_in[13],
                           (const float*)d_in[15] };
  float* out = (float*)d_out;

  const int B = 8192;
  char* ws = (char*)d_ws;
  size_t off = 0;
  auto alloc = [&](size_t bytes) { size_t o = off; off += (bytes + 255) & ~(size_t)255; return o; };

  _Float16* xh = (_Float16*)(ws + alloc((size_t)B * 32 * 32 * 8 * 2)); // ch pad 3->8
  (void)alloc(4096);                                                   // OOB-slack for pad taps
  _Float16* w1 = (_Float16*)(ws + alloc(16 * 224 * 2));
  _Float16* w2 = (_Float16*)(ws + alloc(16 * 224 * 2));
  _Float16* w3 = (_Float16*)(ws + alloc(128 * 400 * 2));
  _Float16* w4 = (_Float16*)(ws + alloc(96 * 128 * 2));
  _Float16* w5 = (_Float16*)(ws + alloc(16 * 96 * 2));
  _Float16* c1 = (_Float16*)(ws + alloc((size_t)B * 784 * 16 * 2)); // conv1 out NHWC(pad16)
  _Float16* p1 = (_Float16*)(ws + alloc((size_t)B * 196 * 16 * 2)); // pool1
  _Float16* a1 = (_Float16*)(ws + alloc((size_t)B * 196 * 8 * 2)); // +-1 act, ch pad 6->8
  (void)alloc(4096);                                                   // OOB-slack for pad taps
  _Float16* c2 = (_Float16*)(ws + alloc((size_t)B * 100 * 16 * 2)); // conv2 out
  _Float16* p2 = (_Float16*)(ws + alloc((size_t)B * 25 * 16 * 2));  // pool2
  _Float16* a2 = (_Float16*)(ws + alloc((size_t)B * 400 * 2));      // flat +-1 (HWC)
  float*    f1 = (float*)   (ws + alloc((size_t)B * 128 * 4));
  _Float16* a3 = (_Float16*)(ws + alloc((size_t)B * 128 * 2));
  float*    f2 = (float*)   (ws + alloc((size_t)B * 96 * 4));
  _Float16* a4 = (_Float16*)(ws + alloc((size_t)B * 96 * 2));
  float*    f3 = (float*)   (ws + alloc((size_t)B * 16 * 4));
  float*    st[5];
  for (int i = 0; i < 5; ++i) st[i] = (float*)(ws + alloc(256 * 4));

  auto g1 = [](long n) { return (unsigned)((n + 255) / 256); };

  // 1) effective binarized weights (conv layouts channel-padded to 8)
  k_make_w<<<g1(16 * 224), 256, 0, stream>>>(conv1_w, w1, 6, 16, 75, 224, 0, 3, 1000u);
  k_make_w<<<g1(16 * 224), 256, 0, stream>>>(conv2_w, w2, 16, 16, 150, 224, 0, 6, 2000u);
  k_make_w<<<g1(128 * 400), 256, 0, stream>>>(fc1_w, w3, 120, 128, 400, 400, 2, 0, 3000u);
  k_make_w<<<g1(96 * 128), 256, 0, stream>>>(fc2_w, w4, 84, 96, 120, 128, 1, 0, 4000u);
  k_make_w<<<g1(16 * 96), 256, 0, stream>>>(fc3_w, w5, 10, 16, 84, 96, 1, 0, 5000u);

  // 2) input NCHW f32 -> NHWC f16, channels padded to 8
  k_x2h<<<g1((long)B * 32 * 32 * 8), 256, 0, stream>>>(x, xh, B * 32 * 32 * 8);

  // 3) conv1: M = B*784, Kpad = 224 (28 taps * 8ch)
  k_gemm_conv<32, 32, 28, 28>
      <<<(unsigned)((size_t)B * 784 / 64), 128, 0, stream>>>(xh, w1, c1);
  k_pool<<<g1((long)B * 196 * 16), 256, 0, stream>>>(c1, p1, B, 28, 28, 16);
  k_stat<_Float16><<<16, 256, 0, stream>>>(p1, st[0], 16, (long)B * 196);
  k_bn_act<_Float16><<<g1((long)B * 196 * 8), 256, 0, stream>>>(
      p1, 16, st[0], bn_g[0], bn_b[0], a1, 8, 6, (long)B * 196,
      1.0f / ((float)B * 196.f), 11u);

  // 4) conv2: M = B*100, Kpad = 224
  k_gemm_conv<14, 14, 10, 10>
      <<<(unsigned)((size_t)B * 100 / 64), 128, 0, stream>>>(a1, w2, c2);
  k_pool<<<g1((long)B * 25 * 16), 256, 0, stream>>>(c2, p2, B, 10, 10, 16);
  k_stat<_Float16><<<16, 256, 0, stream>>>(p2, st[1], 16, (long)B * 25);
  k_bn_act<_Float16><<<g1((long)B * 25 * 16), 256, 0, stream>>>(
      p2, 16, st[1], bn_g[1], bn_b[1], a2, 16, 16, (long)B * 25,
      1.0f / ((float)B * 25.f), 22u);   // a2 == flat [B][400] in HWC order (w3 permuted)

  // 5) fc1: 8192x400 x 400x120(pad128)
  k_gemm_fc<float><<<dim3(B / 64, 8), 128, 0, stream>>>(a2, w3, f1, 400, 400, 128);
  k_stat<float><<<128, 256, 0, stream>>>(f1, st[2], 128, (long)B);
  k_bn_act<float><<<g1((long)B * 128), 256, 0, stream>>>(
      f1, 128, st[2], bn_g[2], bn_b[2], a3, 128, 120, (long)B, 1.0f / (float)B, 33u);

  // 6) fc2: 8192x120(pad128) x 128x84(pad96)
  k_gemm_fc<float><<<dim3(B / 64, 6), 128, 0, stream>>>(a3, w4, f2, 128, 128, 96);
  k_stat<float><<<96, 256, 0, stream>>>(f2, st[3], 96, (long)B);
  k_bn_act<float><<<g1((long)B * 96), 256, 0, stream>>>(
      f2, 96, st[3], bn_g[3], bn_b[3], a4, 96, 84, (long)B, 1.0f / (float)B, 44u);

  // 7) fc3: 8192x84(pad96) x 96x10(pad16), then final BN -> d_out [8192,10] f32
  k_gemm_fc<float><<<dim3(B / 64, 1), 128, 0, stream>>>(a4, w5, f3, 96, 96, 16);
  k_stat<float><<<16, 256, 0, stream>>>(f3, st[4], 16, (long)B);
  k_bn_out<<<g1((long)B * 10), 256, 0, stream>>>(
      f3, 16, st[4], bn_g[4], bn_b[4], out, 10, (long)B, 1.0f / (float)B);
}
